// RNN_3229815407358
// MI455X (gfx1250) — matrix operand order repaired and checked by reading, not compiled, not run on hardware
//
#include <hip/hip_runtime.h>
#include <hip/hip_bf16.h>
#include <math.h>

// ---------------------------------------------------------------------------
// Vector types
// ---------------------------------------------------------------------------
typedef __attribute__((ext_vector_type(16))) __bf16 v16bf;
typedef __attribute__((ext_vector_type(8)))  __bf16 v8bf;
typedef __attribute__((ext_vector_type(4)))  __bf16 v4bf;
typedef __attribute__((ext_vector_type(8)))  float  v8f;
typedef __attribute__((ext_vector_type(4)))  float  v4f;
typedef __attribute__((ext_vector_type(4)))  unsigned v4u;
typedef __attribute__((ext_vector_type(8)))  int    v8i;
typedef __attribute__((ext_vector_type(4)))  int    v4i;

template <typename A, typename B> struct same_type { static constexpr bool value = false; };
template <typename A> struct same_type<A, A>      { static constexpr bool value = true;  };

#if defined(__has_builtin)
#if __has_builtin(__builtin_amdgcn_tensor_load_to_lds) && __has_builtin(__builtin_amdgcn_s_wait_tensorcnt)
#define USE_TDM 1
#endif
#endif

// ---------------------------------------------------------------------------
// Workspace init: zero h double-buffer + grid-barrier state every call so
// graph replays are deterministic (harness poisons d_ws once, never again).
// ---------------------------------------------------------------------------
__global__ void init_ws_kernel(float* __restrict__ hbuf, unsigned* __restrict__ bar) {
    int i = blockIdx.x * blockDim.x + threadIdx.x;
    if (i < 4096) hbuf[i] = 0.0f;          // both h buffers (2 x 2048)
    if (i < 4)    bar[i]  = 0u;            // barrier count + generation
}

// ---------------------------------------------------------------------------
// GEMM with bias:  C[M,N] = A[M,K] @ B[N,K]^T + bias[N]
//   A: f32 or bf16 row-major [M,K];  B: f32 row-major [N,K]  (both K-contig)
// Block = 256 threads = 8 waves. Block tile 64x64, TK=64.
// Wave (wy 0..3, wx 0..1) computes 16 rows x 32 cols = two 16x16 WMMA tiles,
// reusing the A fragment -> 4 v_wmma per K iteration.
// ---------------------------------------------------------------------------
#define GEMM_TM 64
#define GEMM_TN 64
#define GEMM_TK 64
#define LDSS    72      // bf16 elems per LDS row (144 B: 16B-aligned, conflict-free)

// Stage a 64x64 f32 tile -> bf16 LDS tile. Exact trip counts, b128 loads,
// packed bf16 8-byte LDS stores. tid in [0,256).
__device__ __forceinline__ void stage_f32_tile(const float* __restrict__ src, int ld,
                                               __bf16* __restrict__ dst, int tid) {
    #pragma unroll
    for (int i = 0; i < (GEMM_TM * GEMM_TK) / (256 * 4); ++i) {   // 4 iters
        const int f   = tid + i * 256;        // float4 index
        const int row = f >> 4;               // 16 float4 per row
        const int c   = (f & 15) << 2;
        const v4f v = *(const v4f*)(src + (size_t)row * ld + c);
        v4bf b;
        b[0] = (__bf16)v[0]; b[1] = (__bf16)v[1];
        b[2] = (__bf16)v[2]; b[3] = (__bf16)v[3];
        *(v4bf*)(dst + row * LDSS + c) = b;
    }
}

// Stage a 64x64 bf16 tile (manual fallback when no TDM builtin).
__device__ __forceinline__ void stage_bf16_tile(const __bf16* __restrict__ src, int ld,
                                                __bf16* __restrict__ dst, int tid) {
    #pragma unroll
    for (int i = 0; i < (GEMM_TM * GEMM_TK) / (256 * 8); ++i) {   // 2 iters
        const int f   = tid + i * 256;        // 8-elem chunk index
        const int row = f >> 3;               // 8 chunks per row
        const int c   = (f & 7) << 3;
        *(v8bf*)(dst + row * LDSS + c) = *(const v8bf*)(src + (size_t)row * ld + c);
    }
}

// A fragment (16x32 bf16, ISA 7.12.2): lane half g, row r.
__device__ __forceinline__ v16bf load_a_frag(const __bf16* As, int row, int kbase, int g) {
    v16bf fr;
    #pragma unroll
    for (int j = 0; j < 8; ++j) {
        const int ka = kbase + ((j < 4) ? 2 * j : 16 + 2 * (j - 4)) + g * 8;
        fr[2 * j]     = As[row * LDSS + ka];
        fr[2 * j + 1] = As[row * LDSS + ka + 1];
    }
    return fr;
}

// B fragment (32x16 bf16): same K order as A fragment (lane half g: K=8g..8g+7, 16+8g..16+8g+7).
__device__ __forceinline__ v16bf load_b_frag(const __bf16* Bs, int col, int kbase, int g) {
    v16bf fr;
    #pragma unroll
    for (int j = 0; j < 8; ++j) {
        const int kb = kbase + ((j < 4) ? 2 * j : 16 + 2 * (j - 4)) + g * 8;
        fr[2 * j]     = Bs[col * LDSS + kb];
        fr[2 * j + 1] = Bs[col * LDSS + kb + 1];
    }
    return fr;
}

template <typename TA>
__global__ __launch_bounds__(256)
void gemm_bias_wmma(const TA* __restrict__ A, const float* __restrict__ B,
                    const float* __restrict__ bias, float* __restrict__ C,
                    int M, int N, int K) {
    __shared__ __align__(16) __bf16 As[GEMM_TM * LDSS];
    __shared__ __align__(16) __bf16 Bs[GEMM_TN * LDSS];

    const int tid  = threadIdx.x;
    const int wave = tid >> 5;
    const int lane = tid & 31;
    const int wy   = wave >> 1;        // 0..3 : 16-row strip
    const int wx   = wave & 1;         // 0..1 : 32-col strip
    const int g    = lane >> 4;
    const int r    = lane & 15;

    const int m0 = blockIdx.y * GEMM_TM;
    const int n0 = blockIdx.x * GEMM_TN;

    v8f acc0 = {0.f,0.f,0.f,0.f,0.f,0.f,0.f,0.f};
    v8f acc1 = {0.f,0.f,0.f,0.f,0.f,0.f,0.f,0.f};

    for (int k0 = 0; k0 < K; k0 += GEMM_TK) {
        // ---- stage A tile ----
        if constexpr (same_type<TA, float>::value) {
            stage_f32_tile(A + (size_t)m0 * K + k0, K, As, tid);
        } else {
#if defined(USE_TDM)
            // Tensor Data Mover: one wave issues a 2D tile DMA (64x64 bf16),
            // LDS padding 4 DWORDs per 128B row -> row stride 144B (= LDSS).
            if (wave == 0) {
                const unsigned long long ga =
                    (unsigned long long)(const void*)(A + (size_t)m0 * K + k0);
                const unsigned lds_off = (unsigned)(unsigned long long)(const void*)As;
                v4u g0;
                g0[0] = 1u;                                   // count=1 valid descriptor
                g0[1] = lds_off;                              // lds_addr
                g0[2] = (unsigned)ga;                         // global_addr[31:0]
                g0[3] = (unsigned)((ga >> 32) & 0x1FFFFFFu)   // global_addr[56:32]
                      | (2u << 30);                           // type = 2 ("image")
                v8i g1;
                g1[0] = (int)((1u << 16)      // data_size = 1 (2 bytes)
                      |       (1u << 20)      // pad_enable
                      |       (4u << 22)      // pad_interval: 16 x 8B = 128B rows
                      |       (3u << 25));    // pad_amount: 4 DWORDs (16B) -> 144B stride
                g1[1] = (int)(((unsigned)K & 0xFFFFu) << 16);            // tensor_dim0 lo
                g1[2] = (int)(((unsigned)K >> 16) | (((unsigned)M & 0xFFFFu) << 16)); // dim0 hi | dim1 lo
                g1[3] = (int)(((unsigned)M >> 16) | ((unsigned)GEMM_TK << 16));       // dim1 hi | tile_dim0
                g1[4] = (int)(unsigned)GEMM_TM;                          // tile_dim1 | tile_dim2=0
                g1[5] = (int)(unsigned)K;                                // tensor_dim0_stride lo
                g1[6] = 0; g1[7] = 0;
                const v4i gz4 = {0, 0, 0, 0};
                const v8i gz8 = {0, 0, 0, 0, 0, 0, 0, 0};
                __builtin_amdgcn_tensor_load_to_lds(g0, g1, gz4, gz4, gz8, 0);
                __builtin_amdgcn_s_wait_tensorcnt(0);
            }
#else
            stage_bf16_tile(A + (size_t)m0 * K + k0, K, As, tid);
#endif
        }
        // ---- stage B tile (always f32 weights) ----
        stage_f32_tile(B + (size_t)n0 * K + k0, K, Bs, tid);
        __syncthreads();

        // ---- 4 WMMAs: 2 K-halves x 2 N-subtiles, A fragment reused ----
        #pragma unroll
        for (int kh = 0; kh < 2; ++kh) {
            const v16bf af  = load_a_frag(As, wy * 16 + r,      kh * 32, g);
            const v16bf bf0 = load_b_frag(Bs, wx * 32 + r,      kh * 32, g);
            const v16bf bf1 = load_b_frag(Bs, wx * 32 + 16 + r, kh * 32, g);
            acc0 = __builtin_amdgcn_wmma_f32_16x16x32_bf16(
                false, af, false, bf0, (short)0, acc0, false, false);
            acc1 = __builtin_amdgcn_wmma_f32_16x16x32_bf16(
                false, af, false, bf1, (short)0, acc1, false, false);
        }
        __syncthreads();
    }

    // D layout: VGPR i -> row (8g + i), col r within each 16x16 tile.
    const int na = n0 + wx * 32 + r;
    const int nb = na + 16;
    const float bva = bias[na];
    const float bvb = bias[nb];
    #pragma unroll
    for (int i = 0; i < 8; ++i) {
        const int m = m0 + wy * 16 + g * 8 + i;
        C[(size_t)m * N + na] = acc0[i] + bva;
        C[(size_t)m * N + nb] = acc1[i] + bvb;
    }
}

// ---------------------------------------------------------------------------
// Device-wide sense-reversing barrier (agent scope, s_sleep spin)
// ---------------------------------------------------------------------------
__device__ __forceinline__ void grid_barrier(unsigned* cnt, unsigned* gen, unsigned nblk) {
    __syncthreads();
    __threadfence();
    if (threadIdx.x == 0) {
        unsigned g = __hip_atomic_load(gen, __ATOMIC_ACQUIRE, __HIP_MEMORY_SCOPE_AGENT);
        unsigned arrived = __hip_atomic_fetch_add(cnt, 1u, __ATOMIC_ACQ_REL, __HIP_MEMORY_SCOPE_AGENT);
        if (arrived == nblk - 1u) {
            __hip_atomic_store(cnt, 0u, __ATOMIC_RELAXED, __HIP_MEMORY_SCOPE_AGENT);
            __hip_atomic_store(gen, g + 1u, __ATOMIC_RELEASE, __HIP_MEMORY_SCOPE_AGENT);
        } else {
            while (__hip_atomic_load(gen, __ATOMIC_ACQUIRE, __HIP_MEMORY_SCOPE_AGENT) == g)
                __builtin_amdgcn_s_sleep(1);
        }
    }
    __syncthreads();
}

// ---------------------------------------------------------------------------
// Sequential RNN scan: h_t = tanh(W_h h_{t-1} + xp_t), T steps.
// 128 persistent WGs x 256 threads; each WG pins its 16 rows of W_h (128 KB)
// in dynamic LDS (WGP budget 320 KB). Per step only the 8 KB h vector moves
// through L2 plus one device-wide barrier -> latency-bound, not BW-bound.
// Emits hs as bf16 (feeds the W_y WMMA GEMM) and h_final f32.
// ---------------------------------------------------------------------------
#define SCAN_WGS 128
#define SCAN_ROWS 16

__global__ __launch_bounds__(256)
void rnn_scan_kernel(const float* __restrict__ Wh, const float* __restrict__ xp,
                     float* __restrict__ hbuf, __bf16* __restrict__ hs,
                     float* __restrict__ hfinal, unsigned* __restrict__ bar,
                     int T, int H) {
    extern __shared__ float Wl[];            // SCAN_ROWS * H floats (128 KB)
    __shared__ float hsh[2048];
    __shared__ float part[16][17];

    const int tid  = threadIdx.x;
    const int nblk = gridDim.x;
    const int row0 = blockIdx.x * SCAN_ROWS;

    for (int idx = tid; idx < SCAN_ROWS * H; idx += 256)
        Wl[idx] = Wh[(size_t)row0 * H + idx];
    __syncthreads();

    const int ty = tid >> 4;                 // row within slice (0..15)
    const int tx = tid & 15;                 // K chunk          (0..15)

    for (int t = 0; t < T; ++t) {
        const float* hprev = hbuf + (size_t)(t & 1) * H;
        float*       hnext = hbuf + (size_t)((t + 1) & 1) * H;

        for (int i = tid; i < H; i += 256) hsh[i] = hprev[i];
        __syncthreads();

        const float* wr = &Wl[ty * H + tx * 128];
        const float* hv = &hsh[tx * 128];
        float acc = 0.f;
        #pragma unroll 8
        for (int k = 0; k < 128; ++k) acc = __builtin_fmaf(wr[k], hv[k], acc);
        part[ty][tx] = acc;
        __syncthreads();

        if (tx == 0) {
            float s = 0.f;
            #pragma unroll
            for (int j = 0; j < 16; ++j) s += part[ty][j];
            const int row = row0 + ty;
            const float h = tanhf(s + xp[(size_t)t * H + row]);
            hnext[row] = h;
            hs[(size_t)t * H + row] = (__bf16)h;
            if (t == T - 1) hfinal[row] = h;
        }
        grid_barrier(&bar[0], &bar[1], (unsigned)nblk);
    }
}

// ---------------------------------------------------------------------------
// Row softmax: one block per row (N = 2048)
// ---------------------------------------------------------------------------
__global__ __launch_bounds__(256)
void softmax_rows_kernel(const float* __restrict__ logits, float* __restrict__ out, int N) {
    __shared__ float red[256];
    const int row = blockIdx.x;
    const int tid = threadIdx.x;
    const float* in = logits + (size_t)row * N;
    float* o = out + (size_t)row * N;

    float m = -__builtin_inff();
    for (int i = tid; i < N; i += 256) m = fmaxf(m, in[i]);
    red[tid] = m; __syncthreads();
    for (int s = 128; s > 0; s >>= 1) {
        if (tid < s) red[tid] = fmaxf(red[tid], red[tid + s]);
        __syncthreads();
    }
    m = red[0]; __syncthreads();

    float sum = 0.f;
    for (int i = tid; i < N; i += 256) {
        float e = __expf(in[i] - m);
        sum += e;
        o[i] = e;
    }
    red[tid] = sum; __syncthreads();
    for (int s = 128; s > 0; s >>= 1) {
        if (tid < s) red[tid] += red[tid + s];
        __syncthreads();
    }
    const float inv = 1.0f / red[0];
    for (int i = tid; i < N; i += 256) o[i] *= inv;
}

// ---------------------------------------------------------------------------
// Launcher
// ---------------------------------------------------------------------------
extern "C" void kernel_launch(void* const* d_in, const int* in_sizes, int n_in,
                              void* d_out, int out_size, void* d_ws, size_t ws_size,
                              hipStream_t stream) {
    (void)in_sizes; (void)n_in; (void)out_size; (void)ws_size;
    constexpr int T = 8192, E = 2048, H = 2048;

    const float* x  = (const float*)d_in[0];   // [T,E]
    const float* Wh = (const float*)d_in[1];   // [H,H]
    const float* Wx = (const float*)d_in[2];   // [H,E]
    const float* Wy = (const float*)d_in[3];   // [E,H]
    const float* Bh = (const float*)d_in[4];   // [H]
    const float* By = (const float*)d_in[5];   // [E]

    float* out = (float*)d_out;                // [0:H)=h_final, [H:)=softmax

    char* ws = (char*)d_ws;
    size_t off = 0;
    float*  xp   = (float*)(ws + off);  off += (size_t)T * H * sizeof(float);   // reused as logits
    __bf16* hs   = (__bf16*)(ws + off); off += (size_t)T * H * sizeof(__bf16);
    float*  hbuf = (float*)(ws + off);  off += (size_t)2 * H * sizeof(float) + 8192;
    unsigned* bar = (unsigned*)(ws + off);

    // 0) deterministic per-call init (h0 = 0, barrier state = 0)
    init_ws_kernel<<<16, 256, 0, stream>>>(hbuf, bar);

    // 1) xp = x @ Wx^T + Bh   (WMMA bf16, f32 accum)
    gemm_bias_wmma<float><<<dim3(H / GEMM_TN, T / GEMM_TM), 256, 0, stream>>>(
        x, Wx, Bh, xp, T, H, E);

    // 2) sequential scan (persistent, W_h slice pinned in LDS)
    rnn_scan_kernel<<<SCAN_WGS, 256, SCAN_ROWS * H * sizeof(float), stream>>>(
        Wh, xp, hbuf, hs, out, bar, T, H);

    // 3) logits = hs @ Wy^T + By   (bf16 A via Tensor Data Mover when available)
    gemm_bias_wmma<__bf16><<<dim3(E / GEMM_TN, T / GEMM_TM), 256, 0, stream>>>(
        hs, Wy, By, xp, T, E, H);

    // 4) row softmax -> d_out
    softmax_rows_kernel<<<T, 256, 0, stream>>>(xp, out + H, E);
}
